// GATv2_9053791060057
// MI455X (gfx1250) — compile-verified, matrix operand order checked by reading, not run on hardware
//
#include <hip/hip_runtime.h>
#include <hip/hip_bf16.h>

#define NNODES 50000
#define NEDGES 1600000
#define NTOT   1650000   /* E + N self loops */
#define HC 128
#define NH 4
#define NOUT 16

typedef __attribute__((ext_vector_type(16))) _Float16 v16h;
typedef __attribute__((ext_vector_type(8)))  _Float16 v8h;
typedef __attribute__((ext_vector_type(8)))  float    v8f;

// ---- order-preserving float<->uint encoding for atomic max on floats ----
__device__ __forceinline__ unsigned fenc(float f) {
    unsigned u = __float_as_uint(f);
    return (u & 0x80000000u) ? ~u : (u | 0x80000000u);
}
__device__ __forceinline__ float fdec(unsigned e) {
    unsigned b = (e & 0x80000000u) ? (e & 0x7FFFFFFFu) : ~e;
    return __uint_as_float(b);
}

// ---------------- init / convert kernels ----------------
__global__ void zero_f(float* __restrict__ p, int n) {
    int i = blockIdx.x * blockDim.x + threadIdx.x;
    if (i < n) p[i] = 0.f;
}
__global__ void fill_u(unsigned* __restrict__ p, int n, unsigned v) {
    int i = blockIdx.x * blockDim.x + threadIdx.x;
    if (i < n) p[i] = v;
}
__global__ void cvt_f32_f16(const float* __restrict__ src, _Float16* __restrict__ dst, int n) {
    int i = blockIdx.x * blockDim.x + threadIdx.x;
    if (i < n) dst[i] = (_Float16)src[i];
}
// transpose-convert weights: Wt[n*128 + k] = (f16) W[k*ncols + n]   (K always 128)
__global__ void wt_cvt(const float* __restrict__ W, _Float16* __restrict__ Wt, int ncols) {
    int t = blockIdx.x * blockDim.x + threadIdx.x;
    if (t >= 128 * ncols) return;
    int n = t >> 7;          // t / 128
    int k = t & 127;         // t % 128
    Wt[t] = (_Float16)W[k * ncols + n];
}

// ---------------- self-loop attribute: mean incoming edge weight ----------------
__global__ void deg_kernel(const long long* __restrict__ ei, const float* __restrict__ ew,
                           float* __restrict__ dsum, float* __restrict__ dcnt) {
    int e = blockIdx.x * blockDim.x + threadIdx.x;
    if (e >= NEDGES) return;
    int dst = (int)ei[NEDGES + e];
    atomicAdd(&dsum[dst], ew[e]);
    atomicAdd(&dcnt[dst], 1.0f);
}
__global__ void lattr_kernel(const float* __restrict__ dsum, const float* __restrict__ dcnt,
                             float* __restrict__ lattr) {
    int i = blockIdx.x * blockDim.x + threadIdx.x;
    if (i >= NNODES) return;
    lattr[i] = dsum[i] / fmaxf(dcnt[i], 1.0f);
}

// ---------------- WMMA GEMM:  Y[N x NC] = A16[N x 128] @ Wt16^T + bias ----------------
// A16: f16 row-major (row stride 128). Wt16: f16, weight TRANSPOSED (Wt[n*128+k]).
// One wave32 per 16x16 tile, K=128 in 4 chunks of 32. Per chunk each operand is two
// contiguous 16B runs per lane (ISA 16-bit A layout: lane<16 K={0..7,16..23}, lane>=16 +8),
// so the whole inner loop is 4x global_load_b128 + 1x v_wmma_f32_16x16x32_f16.
template <int NC>
__global__ void gemm_wmma(const _Float16* __restrict__ A, const _Float16* __restrict__ Wt,
                          const float* __restrict__ bias, float* __restrict__ Y) {
    const int lane = threadIdx.x;          // 0..31
    const int n0   = threadIdx.y * 16;     // col tile within block
    const int m0   = blockIdx.x * 16;      // row tile (N divides by 16)
    const int half = lane >> 4;            // 0 or 1
    const int l16  = lane & 15;
    const int kb   = half * 8;
    const size_t arow = (size_t)(m0 + l16) * 128;
    const size_t brow = (size_t)(n0 + l16) * 128;

    v8f c = {};
#pragma unroll
    for (int kk = 0; kk < 128; kk += 32) {
        const v8h a0 = *(const v8h*)(A + arow + kk + kb);
        const v8h a1 = *(const v8h*)(A + arow + kk + kb + 16);
        const v8h b0 = *(const v8h*)(Wt + brow + kk + kb);
        const v8h b1 = *(const v8h*)(Wt + brow + kk + kb + 16);
        const v16h a = __builtin_shufflevector(a0, a1, 0,1,2,3,4,5,6,7,8,9,10,11,12,13,14,15);
        const v16h b = __builtin_shufflevector(b0, b1, 0,1,2,3,4,5,6,7,8,9,10,11,12,13,14,15);
        c = __builtin_amdgcn_wmma_f32_16x16x32_f16(
                /*neg_a=*/false, a, /*neg_b=*/false, b,
                /*c_mod=*/(short)0, c, /*reuse_a=*/false, /*reuse_b=*/false);
    }
#pragma unroll
    for (int r = 0; r < 8; ++r) {
        const int m = m0 + r + half * 8;   // C/D layout: VGPR r -> M=r (lanes 0-15) / r+8
        const int n = n0 + l16;
        Y[(size_t)m * NC + n] = c[r] + bias[n];
    }
}

// ---------------- pass 1: attention logits + segment max ----------------
// One wave32 per edge; 4 channels/lane (float4 gathers stay L2-resident).
__global__ void edge_alpha(const long long* __restrict__ ei, const float* __restrict__ ew,
                           const float* __restrict__ lattr,
                           const float* __restrict__ xl, const float* __restrict__ xr,
                           const float* __restrict__ We, const float* __restrict__ att,
                           float* __restrict__ alpha, unsigned* __restrict__ amax) {
    const int lane = threadIdx.x & 31;
    const int e = (blockIdx.x * blockDim.x + threadIdx.x) >> 5;
    if (e >= NTOT) return;
    int src, dst; float w;
    if (e < NEDGES) { src = (int)ei[e]; dst = (int)ei[NEDGES + e]; w = ew[e]; }
    else            { src = dst = e - NEDGES; w = lattr[src]; }

    const int c0 = lane * 4;
    const float4 vl = *(const float4*)(xl + (size_t)src * HC + c0);
    const float4 vr = *(const float4*)(xr + (size_t)dst * HC + c0);
    const float4 ve = *(const float4*)(We + c0);
    const float4 va = *(const float4*)(att + c0);

    float p = 0.f, m;
    m = vl.x + vr.x + w * ve.x; m = (m > 0.f) ? m : 0.2f * m; p += m * va.x;
    m = vl.y + vr.y + w * ve.y; m = (m > 0.f) ? m : 0.2f * m; p += m * va.y;
    m = vl.z + vr.z + w * ve.z; m = (m > 0.f) ? m : 0.2f * m; p += m * va.z;
    m = vl.w + vr.w + w * ve.w; m = (m > 0.f) ? m : 0.2f * m; p += m * va.w;

    // reduce within aligned 8-lane head groups (wave32)
    p += __shfl_xor(p, 1);
    p += __shfl_xor(p, 2);
    p += __shfl_xor(p, 4);

    if ((lane & 7) == 0) {
        const int h = lane >> 3;
        alpha[(size_t)e * NH + h] = p;
        atomicMax(&amax[(size_t)dst * NH + h], fenc(p));
    }
}

// ---------------- pass 2: exp(alpha - max) + segment sum ----------------
__global__ void edge_exp(const long long* __restrict__ ei, float* __restrict__ alpha,
                         const unsigned* __restrict__ amax, float* __restrict__ den) {
    const int t = blockIdx.x * blockDim.x + threadIdx.x;
    if (t >= NTOT * NH) return;
    const int e = t >> 2;
    const int h = t & 3;
    const int dst = (e < NEDGES) ? (int)ei[NEDGES + e] : (e - NEDGES);
    const float ex = __expf(alpha[t] - fdec(amax[(size_t)dst * NH + h]));
    alpha[t] = ex;
    atomicAdd(&den[(size_t)dst * NH + h], ex);
}

// ---------------- pass 3: normalized weighted aggregation ----------------
__global__ void edge_aggr(const long long* __restrict__ ei, const float* __restrict__ alpha,
                          const float* __restrict__ den, const float* __restrict__ xl,
                          float* __restrict__ agg) {
    const int lane = threadIdx.x & 31;
    const int e = (blockIdx.x * blockDim.x + threadIdx.x) >> 5;
    if (e >= NTOT) return;
    int src, dst;
    if (e < NEDGES) { src = (int)ei[e]; dst = (int)ei[NEDGES + e]; }
    else            { src = dst = e - NEDGES; }

    const int h = lane >> 3;
    const float a = alpha[(size_t)e * NH + h] / (den[(size_t)dst * NH + h] + 1e-16f);
    const int c0 = lane * 4;
    const float4 vl = *(const float4*)(xl + (size_t)src * HC + c0);
    float* o = agg + (size_t)dst * HC + c0;
    atomicAdd(o + 0, a * vl.x);
    atomicAdd(o + 1, a * vl.y);
    atomicAdd(o + 2, a * vl.z);
    atomicAdd(o + 3, a * vl.w);
}

// ---------------- bias + ELU, emitting f16 activation for the next GEMM ----------------
__global__ void bias_elu_f16(const float* __restrict__ hbuf, const float* __restrict__ bias,
                             _Float16* __restrict__ a16, int n) {
    const int i = blockIdx.x * blockDim.x + threadIdx.x;
    if (i >= n) return;
    float v = hbuf[i] + bias[i & (HC - 1)];
    v = (v > 0.f) ? v : (__expf(v) - 1.f);
    a16[i] = (_Float16)v;
}

extern "C" void kernel_launch(void* const* d_in, const int* in_sizes, int n_in,
                              void* d_out, int out_size, void* d_ws, size_t ws_size,
                              hipStream_t stream) {
    (void)in_sizes; (void)n_in; (void)out_size; (void)ws_size;
    const float*     x     = (const float*)d_in[0];
    const long long* ei    = (const long long*)d_in[1];
    const float*     ew    = (const float*)d_in[2];
    const float*     Wl1   = (const float*)d_in[3];
    const float*     bl1   = (const float*)d_in[4];
    const float*     Wr1   = (const float*)d_in[5];
    const float*     br1   = (const float*)d_in[6];
    const float*     We1   = (const float*)d_in[7];
    const float*     att1  = (const float*)d_in[8];
    const float*     bias1 = (const float*)d_in[9];
    const float*     Wl2   = (const float*)d_in[10];
    const float*     bl2   = (const float*)d_in[11];
    const float*     Wr2   = (const float*)d_in[12];
    const float*     br2   = (const float*)d_in[13];
    const float*     We2   = (const float*)d_in[14];
    const float*     att2  = (const float*)d_in[15];
    const float*     bias2 = (const float*)d_in[16];
    const float*     Wlin  = (const float*)d_in[17];
    const float*     blin  = (const float*)d_in[18];
    float*           out   = (float*)d_out;

    // workspace layout (node tables xl/xr/hb total 77MB -> L2 resident)
    float*     ws    = (float*)d_ws;
    float*     xl    = ws;                                     // N*HC f32
    float*     xr    = xl + (size_t)NNODES * HC;               // N*HC f32
    float*     hb    = xr + (size_t)NNODES * HC;               // N*HC f32 (agg target)
    float*     alpha = hb + (size_t)NNODES * HC;               // NTOT*NH f32
    unsigned*  amax  = (unsigned*)(alpha + (size_t)NTOT * NH); // N*NH u32
    float*     den   = (float*)(amax + (size_t)NNODES * NH);   // N*NH f32
    float*     dsum  = den + (size_t)NNODES * NH;              // N
    float*     dcnt  = dsum + NNODES;                          // N
    float*     lattr = dcnt + NNODES;                          // N
    _Float16*  a16   = (_Float16*)(lattr + NNODES);            // N*HC f16 (GEMM activations)
    _Float16*  wt    = a16 + (size_t)NNODES * HC;              // 128*128 f16 (weight staging)

    const int TB = 256;
    auto cdiv = [](long long a, long long b) { return (int)((a + b - 1) / b); };
    const unsigned ENC_NEG_INF = 0x007FFFFFu;   // fenc(-inf)
    const dim3 gemmBlk(32, 8);                  // 8 waves: one per 16-col tile
    const int  gemmGrid = NNODES / 16;          // 3125, exact
    const int  edgeBlocks = NTOT / 8;           // 8 waves/block, exact
    const int  nHC = NNODES * HC;

    // self-loop attributes
    zero_f<<<cdiv(NNODES, TB), TB, 0, stream>>>(dsum, NNODES);
    zero_f<<<cdiv(NNODES, TB), TB, 0, stream>>>(dcnt, NNODES);
    deg_kernel<<<cdiv(NEDGES, TB), TB, 0, stream>>>(ei, ew, dsum, dcnt);
    lattr_kernel<<<cdiv(NNODES, TB), TB, 0, stream>>>(dsum, dcnt, lattr);

    // ---- layer 1 ----
    cvt_f32_f16<<<cdiv(nHC, TB), TB, 0, stream>>>(x, a16, nHC);
    wt_cvt<<<cdiv(128 * HC, TB), TB, 0, stream>>>(Wl1, wt, HC);
    gemm_wmma<HC><<<gemmGrid, gemmBlk, 0, stream>>>(a16, wt, bl1, xl);
    wt_cvt<<<cdiv(128 * HC, TB), TB, 0, stream>>>(Wr1, wt, HC);
    gemm_wmma<HC><<<gemmGrid, gemmBlk, 0, stream>>>(a16, wt, br1, xr);

    fill_u<<<cdiv(NNODES * NH, TB), TB, 0, stream>>>(amax, NNODES * NH, ENC_NEG_INF);
    zero_f<<<cdiv(NNODES * NH, TB), TB, 0, stream>>>(den, NNODES * NH);
    zero_f<<<cdiv(nHC, TB), TB, 0, stream>>>(hb, nHC);
    edge_alpha<<<edgeBlocks, TB, 0, stream>>>(ei, ew, lattr, xl, xr, We1, att1, alpha, amax);
    edge_exp<<<cdiv((long long)NTOT * NH, TB), TB, 0, stream>>>(ei, alpha, amax, den);
    edge_aggr<<<edgeBlocks, TB, 0, stream>>>(ei, alpha, den, xl, hb);
    bias_elu_f16<<<cdiv(nHC, TB), TB, 0, stream>>>(hb, bias1, a16, nHC);

    // ---- layer 2 ----
    wt_cvt<<<cdiv(128 * HC, TB), TB, 0, stream>>>(Wl2, wt, HC);
    gemm_wmma<HC><<<gemmGrid, gemmBlk, 0, stream>>>(a16, wt, bl2, xl);
    wt_cvt<<<cdiv(128 * HC, TB), TB, 0, stream>>>(Wr2, wt, HC);
    gemm_wmma<HC><<<gemmGrid, gemmBlk, 0, stream>>>(a16, wt, br2, xr);

    fill_u<<<cdiv(NNODES * NH, TB), TB, 0, stream>>>(amax, NNODES * NH, ENC_NEG_INF);
    zero_f<<<cdiv(NNODES * NH, TB), TB, 0, stream>>>(den, NNODES * NH);
    zero_f<<<cdiv(nHC, TB), TB, 0, stream>>>(hb, nHC);
    edge_alpha<<<edgeBlocks, TB, 0, stream>>>(ei, ew, lattr, xl, xr, We2, att2, alpha, amax);
    edge_exp<<<cdiv((long long)NTOT * NH, TB), TB, 0, stream>>>(ei, alpha, amax, den);
    edge_aggr<<<edgeBlocks, TB, 0, stream>>>(ei, alpha, den, xl, hb);
    bias_elu_f16<<<cdiv(nHC, TB), TB, 0, stream>>>(hb, bias2, a16, nHC);

    // ---- final linear 128 -> 16 ----
    wt_cvt<<<cdiv(128 * NOUT, TB), TB, 0, stream>>>(Wlin, wt, NOUT);
    gemm_wmma<NOUT><<<gemmGrid, dim3(32, 1), 0, stream>>>(a16, wt, blin, out);
}
